// SpiralEnblock_37391985279448
// MI455X (gfx1250) — matrix operand order repaired and checked by reading, not compiled, not run on hardware
//
#include <hip/hip_runtime.h>
#include <stdint.h>

typedef __attribute__((ext_vector_type(16))) __bf16 v16bf;
typedef __attribute__((ext_vector_type(8)))  float  v8f;

constexpr int B_   = 16;
constexpr int N_   = 16384;
constexpr int CIN  = 64;
constexpr int S_   = 16;
constexpr int COUT = 128;
constexpr int M_   = 4096;
constexpr int NNZ  = 3 * M_;          // 12288
constexpr int K_   = CIN * S_;        // 1024

constexpr int TILE_ROWS = 128;        // nnz rows per workgroup
constexpr int KCHUNK    = 128;        // K staged per iteration (2 spiral steps)
constexpr int A_STRIDE  = 136;        // padded bf16 row stride (272 B -> conflict-free b128)
constexpr int W_STRIDE  = 136;

union Frag { uint4 u[2]; v16bf v; };

// ---- f32 -> bf16 (RNE) conversion into workspace -------------------------
__global__ void cvt_bf16_kernel(const float* __restrict__ src,
                                uint16_t* __restrict__ dst, int n) {
  int i = blockIdx.x * blockDim.x + threadIdx.x;
  if (i < n) {
    uint32_t u = __float_as_uint(src[i]);
    uint32_t r = (u + 0x7FFFu + ((u >> 16) & 1u)) >> 16;
    dst[i] = (uint16_t)r;
  }
}

// ---- zero the pooled output ----------------------------------------------
__global__ void zero_kernel(float* __restrict__ p, int n) {
  int i = blockIdx.x * blockDim.x + threadIdx.x;
  if (i < n) p[i] = 0.0f;
}

// ---- fused gather + WMMA GEMM + ELU + weighted scatter-add ---------------
__launch_bounds__(256)
__global__ void spiral_fused_kernel(const uint16_t* __restrict__ xbf,     // [B,N,CIN] bf16
                                    const uint16_t* __restrict__ wbf,     // [COUT,K] bf16
                                    const float*    __restrict__ bias,    // [COUT]
                                    const int*      __restrict__ indices, // [N,S]
                                    const int*      __restrict__ trow,    // [NNZ]
                                    const int*      __restrict__ tcol,    // [NNZ]
                                    const float*    __restrict__ tval,    // [NNZ]
                                    float*          __restrict__ pooled)  // [B,M,COUT]
{
  __shared__ __align__(16) uint16_t sA[TILE_ROWS * A_STRIDE]; // 34816 B
  __shared__ __align__(16) uint16_t sW[COUT * W_STRIDE];      // 34816 B
  __shared__ int   sIdx[TILE_ROWS * S_];                      // 8192 B
  __shared__ int   sRow[TILE_ROWS];
  __shared__ float sVal[TILE_ROWS];

  const int b    = blockIdx.y;
  const int k0   = blockIdx.x * TILE_ROWS;           // first nnz of this tile
  const int tid  = threadIdx.x;
  const int lane = tid & 31;
  const int wave = tid >> 5;
  const int wr   = wave & 3;   // 32-row slab
  const int wc   = wave >> 2;  // 64-col slab

  // Stage nnz metadata and pre-gather all spiral indices for the 128 rows.
  for (int l = tid; l < TILE_ROWS * S_; l += 256) {
    int r = l >> 4, s = l & 15;
    sIdx[l] = indices[tcol[k0 + r] * S_ + s];
  }
  for (int i = tid; i < TILE_ROWS; i += 256) {
    sRow[i] = trow[k0 + i];
    sVal[i] = tval[k0 + i];
  }

  // 8 f32 accumulators: 32 rows x 64 cols per wave (2 row-tiles x 4 col-tiles)
  v8f acc[8] = {};

  const uint16_t* xb = xbf + (size_t)b * N_ * CIN;

  for (int kk = 0; kk < K_ / KCHUNK; ++kk) {
    __syncthreads();
    // ---- stage A: 128 rows x 128 K (gathered spiral rows), bf16 ----
    #pragma unroll
    for (int it = 0; it < 8; ++it) {
      int l  = it * 256 + tid;        // 0..2047, one uint4 (8 bf16) each
      int r  = l >> 4;
      int c8 = (l & 15) * 8;          // bf16 offset within 128-wide chunk
      int s  = kk * 2 + (c8 >> 6);    // spiral step
      int cg = c8 & (CIN - 1);        // channel within x row
      const uint16_t* src = xb + (size_t)sIdx[r * S_ + s] * CIN + cg;
      *reinterpret_cast<uint4*>(&sA[r * A_STRIDE + c8]) =
          *reinterpret_cast<const uint4*>(src);
    }
    // ---- stage W slice: 128 out-ch x 128 K, bf16 ----
    #pragma unroll
    for (int it = 0; it < 8; ++it) {
      int l  = it * 256 + tid;        // 0..2047
      int o  = l >> 4;
      int c8 = (l & 15) * 8;
      *reinterpret_cast<uint4*>(&sW[o * W_STRIDE + c8]) =
          *reinterpret_cast<const uint4*>(wbf + (size_t)o * K_ + kk * KCHUNK + c8);
    }
    __syncthreads();

    // ---- 4 x K=32 WMMA steps over this chunk ----
    #pragma unroll
    for (int ks = 0; ks < 4; ++ks) {
      const int kbase = ks * 32;
      // A fragments (16x32 bf16): lanes 0-15 -> K {0..7,16..23}, lanes 16-31 -> +8
      const int abase = kbase + ((lane >> 4) << 3);
      Frag af[2];
      #pragma unroll
      for (int rr = 0; rr < 2; ++rr) {
        const int ar = wr * 32 + rr * 16 + (lane & 15);
        af[rr].u[0] = *reinterpret_cast<const uint4*>(&sA[ar * A_STRIDE + abase]);
        af[rr].u[1] = *reinterpret_cast<const uint4*>(&sA[ar * A_STRIDE + abase + 16]);
      }
      #pragma unroll
      for (int t = 0; t < 4; ++t) {
        // B fragment (32x16 bf16): lane -> column, K {0..7,16..23}, lanes 16-31 -> +8
        const int o  = wc * 64 + t * 16 + (lane & 15);
        const int kb = kbase + ((lane >> 4) << 3);
        Frag bf;
        bf.u[0] = *reinterpret_cast<const uint4*>(&sW[o * W_STRIDE + kb]);
        bf.u[1] = *reinterpret_cast<const uint4*>(&sW[o * W_STRIDE + kb + 16]);
        #pragma unroll
        for (int rr = 0; rr < 2; ++rr) {
          acc[rr * 4 + t] = __builtin_amdgcn_wmma_f32_16x16x32_bf16(
              false, af[rr].v, false, bf.v, (short)0, acc[rr * 4 + t],
              false, false);
        }
      }
    }
  }
  __syncthreads();

  // ---- epilogue: bias + ELU + per-row scale + atomic scatter-add ----
  const int hi   = lane >> 4;
  const int ncol = lane & 15;
  #pragma unroll
  for (int t = 0; t < 4; ++t) {
    const int col = wc * 64 + t * 16 + ncol;
    const float bc = bias[col];
    #pragma unroll
    for (int rr = 0; rr < 2; ++rr) {
      #pragma unroll
      for (int v = 0; v < 8; ++v) {
        const int rloc = wr * 32 + rr * 16 + hi * 8 + v; // C/D: VGPR v -> M=v (+8 hi)
        float o = acc[rr * 4 + t][v] + bc;
        o = (o > 0.0f) ? o : (__expf(o) - 1.0f);         // ELU
        const float contrib = o * sVal[rloc];
        float* dst = pooled + ((size_t)b * M_ + sRow[rloc]) * COUT + col;
        __hip_atomic_fetch_add(dst, contrib, __ATOMIC_RELAXED,
                               __HIP_MEMORY_SCOPE_AGENT);
      }
    }
  }
}

extern "C" void kernel_launch(void* const* d_in, const int* in_sizes, int n_in,
                              void* d_out, int out_size, void* d_ws, size_t ws_size,
                              hipStream_t stream) {
  const float* x       = (const float*)d_in[0];
  const float* W       = (const float*)d_in[1];
  const float* bias    = (const float*)d_in[2];
  const int*   indices = (const int*)d_in[3];
  const int*   trowp   = (const int*)d_in[4];
  const int*   tcolp   = (const int*)d_in[5];
  const float* tvalp   = (const float*)d_in[6];
  float*       pooled  = (float*)d_out;

  const int n_x = B_ * N_ * CIN;   // 16,777,216
  const int n_w = COUT * K_;       // 131,072

  uint16_t* xbf = (uint16_t*)d_ws;
  uint16_t* wbf = (uint16_t*)((char*)d_ws + (size_t)n_x * sizeof(uint16_t));

  // 1) convert x and W to bf16 in workspace (both then live in L2)
  cvt_bf16_kernel<<<(n_x + 255) / 256, 256, 0, stream>>>(x, xbf, n_x);
  cvt_bf16_kernel<<<(n_w + 255) / 256, 256, 0, stream>>>(W, wbf, n_w);

  // 2) zero the pooled output (scatter-add target)
  zero_kernel<<<(out_size + 255) / 256, 256, 0, stream>>>(pooled, out_size);

  // 3) fused gather + WMMA GEMM + ELU + weighted scatter-add
  dim3 grid(NNZ / TILE_ROWS, B_);  // (96, 16)
  spiral_fused_kernel<<<grid, 256, 0, stream>>>(xbf, wbf, bias, indices,
                                                trowp, tcolp, tvalp, pooled);
}
